// BranchedAttention_52458730553975
// MI455X (gfx1250) — matrix operand order repaired and checked by reading, not compiled, not run on hardware
//
#include <hip/hip_runtime.h>
#include <hip/hip_bf16.h>

// ---------- WMMA types ----------
typedef __attribute__((ext_vector_type(16))) __bf16        v16bf;
typedef __attribute__((ext_vector_type(8)))  float         v8f;
typedef __attribute__((ext_vector_type(4)))  unsigned int  v4u;

union Frag {                    // 32 bytes: one WMMA 16-bit operand fragment
    v16bf        v;
    v4u          q[2];
    unsigned int d[8];
    unsigned short u[16];
};

__device__ __forceinline__ unsigned short f2bf(float x) {
    union { float f; unsigned int u; } c; c.f = x;
    unsigned int r = c.u + 0x7FFFu + ((c.u >> 16) & 1u);   // RNE
    return (unsigned short)(r >> 16);
}

__device__ __forceinline__ v8f vzero8() {
    v8f z;
#pragma unroll
    for (int e = 0; e < 8; ++e) z[e] = 0.0f;
    return z;
}

// ---------------- constants ----------------
#define NH 8
#define HSZ 1024
#define DK 128
#define BQ 4096                         // B*Q == B*S rows
#define QSCALE 0.08838834764831845f     // 1/sqrt(128)
#define SELU_A 1.6732632423543772f
#define SELU_S 1.0507009873554805f

// Register-blocked bf16 GEMM k-loop:
//   C[MT][NT] (16x16 f32 tiles) += A((MT*16)xK) * B(Kx(NT*16))
// A: row-major, K contiguous, stride lda.  Bt: N-major (Bt[n][k]), stride ldb.
// Per k-step: MT+NT x (2 b128) loads feed MT*NT WMMAs (register reuse).
template <int MT, int NT, bool PFA, bool PFB>
__device__ __forceinline__ void wmma_block_kloop(v8f (&acc)[MT][NT],
        const unsigned short* __restrict__ A,  int lda,
        const unsigned short* __restrict__ Bt, int ldb, int K)
{
    const int lane = threadIdx.x & 31;
    const int half = lane >> 4;          // 0: lanes 0-15, 1: lanes 16-31
    const int mn   = lane & 15;
    const unsigned short* pa[MT];
    const unsigned short* pb[NT];
#pragma unroll
    for (int mi = 0; mi < MT; ++mi) pa[mi] = A  + (long)(mi * 16 + mn) * lda + 8  * half;
#pragma unroll
    for (int ni = 0; ni < NT; ++ni) pb[ni] = Bt + (long)(ni * 16 + mn) * ldb + 8  * half;

    for (int k0 = 0; k0 < K; k0 += 32) {
        Frag a[MT], b[NT];
#pragma unroll
        for (int mi = 0; mi < MT; ++mi) {
            a[mi].q[0] = *(const v4u*)(pa[mi] + k0);        // K = 8h .. 8h+7
            a[mi].q[1] = *(const v4u*)(pa[mi] + k0 + 16);   // K = 16+8h .. +7
            if (PFA) __builtin_prefetch(pa[mi] + k0 + 128, 0, 3);
        }
#pragma unroll
        for (int ni = 0; ni < NT; ++ni) {
            b[ni].q[0] = *(const v4u*)(pb[ni] + k0);        // K = 8h .. 8h+7
            b[ni].q[1] = *(const v4u*)(pb[ni] + k0 + 16);   // K = 16+8h .. +7
            if (PFB) __builtin_prefetch(pb[ni] + k0 + 128, 0, 3);
        }
#pragma unroll
        for (int mi = 0; mi < MT; ++mi)
#pragma unroll
            for (int ni = 0; ni < NT; ++ni)
                acc[mi][ni] = __builtin_amdgcn_wmma_f32_16x16x32_bf16(
                    false, a[mi].v, false, b[ni].v, (short)0, acc[mi][ni],
                    false, false);
    }
}

// ============ prep 1: flat f32 -> bf16 convert (8 elems/thread, b128 I/O) ===
__global__ __launch_bounds__(256)
void convert_bf16_kernel(const float4* __restrict__ src, v4u* __restrict__ dst,
                         long n8)
{
    const long i = (long)blockIdx.x * blockDim.x + threadIdx.x;
    if (i >= n8) return;
    const float4 a = src[i * 2], b = src[i * 2 + 1];
    Frag t;
    t.u[0] = f2bf(a.x); t.u[1] = f2bf(a.y); t.u[2] = f2bf(a.z); t.u[3] = f2bf(a.w);
    t.u[4] = f2bf(b.x); t.u[5] = f2bf(b.y); t.u[6] = f2bf(b.z); t.u[7] = f2bf(b.w);
    dst[i] = t.q[0];
}

// ============ prep 2: batched transpose + convert: dst[b][c][r] = src[b][r][c]
__global__ __launch_bounds__(256)
void transpose_bf16_kernel(const float* __restrict__ src,
                           unsigned short* __restrict__ dst,
                           int R, int C, long total)
{
    const long idx = ((long)blockIdx.x * blockDim.x + threadIdx.x) * 2;
    if (idx >= total) return;
    const int  r = (int)(idx % R);          // even (R even, idx even)
    const long t = idx / R;
    const int  c = (int)(t % C);
    const long b = t / C;
    const float* s = src + (b * R + r) * (long)C + c;
    const unsigned int lo = f2bf(s[0]);
    const unsigned int hi = f2bf(s[C]);
    *(unsigned int*)(dst + idx) = lo | (hi << 16);
}

// ============ Kernel 1: fused Q/K/V per-head projections ====================
// grid (64 m-tiles of 64 rows, 8 heads, 3 {q,k,v}); block 256 = 8 waves, each
// wave one 32x32 (2x2 WMMA) tile.  [4096x1024]x[1024x128] bf16.
__global__ __launch_bounds__(256)
void qkv_proj_kernel(const unsigned short* __restrict__ xin,   // [3][4096][1024]
                     const unsigned short* __restrict__ wt,    // [3][8][128][1024]
                     unsigned short* __restrict__ qh,          // [8][4096][128]
                     unsigned short* __restrict__ kh,          // [8][4096][128]
                     unsigned short* __restrict__ vhT)         // [8][4][128][1024]
{
    const int n = blockIdx.y, z = blockIdx.z, mt = blockIdx.x;
    const int wave = threadIdx.x >> 5, lane = threadIdx.x & 31;
    const int half = lane >> 4, nn = lane & 15;
    const int r0 = mt * 64 + (wave >> 2) * 32;        // row base (32 rows)
    const int c0 = (wave & 3) * 32;                   // col base (32 of 128)

    const unsigned short* X  = xin + (long)z * BQ * HSZ + (long)r0 * HSZ;
    const unsigned short* Bt = wt + ((long)(z * NH + n) * DK + c0) * HSZ;

    v8f acc[2][2];
#pragma unroll
    for (int mi = 0; mi < 2; ++mi)
#pragma unroll
        for (int ni = 0; ni < 2; ++ni) acc[mi][ni] = vzero8();

    wmma_block_kloop<2, 2, true, true>(acc, X, HSZ, Bt, HSZ, HSZ);

    if (z == 2) {
        // V: store transposed [d][s]; 8 consecutive bf16 per lane -> one b128
        const int b = r0 >> 10, s0 = r0 & 1023;
#pragma unroll
        for (int mi = 0; mi < 2; ++mi)
#pragma unroll
            for (int ni = 0; ni < 2; ++ni) {
                Frag t;
#pragma unroll
                for (int r = 0; r < 8; ++r) t.u[r] = f2bf(acc[mi][ni][r]);
                unsigned short* dst = vhT
                    + ((long)(n * 4 + b) * DK + c0 + ni * 16 + nn) * 1024
                    + s0 + mi * 16 + 8 * half;
                *(v4u*)dst = t.q[0];
            }
    } else {
        const float scale = (z == 0) ? QSCALE : 1.0f;  // fold 1/sqrt(dk) into qh
        unsigned short* Y = ((z == 0) ? qh : kh) + (long)n * BQ * DK;
#pragma unroll
        for (int mi = 0; mi < 2; ++mi)
#pragma unroll
            for (int ni = 0; ni < 2; ++ni)
#pragma unroll
                for (int r = 0; r < 8; ++r)
                    Y[(long)(r0 + mi * 16 + r + 8 * half) * DK + c0 + ni * 16 + nn]
                        = f2bf(acc[mi][ni][r] * scale);
    }
}

// ============ Kernel 2: scores + softmax + attn-out =========================
// grid (64 q-tiles, B=4, NH=8); block 256.  16 q-rows x S=1024 scores in LDS,
// softmax in place, probs -> global attn (f32) and bf16 LDS for P@V WMMA.
__global__ __launch_bounds__(256)
void attn_kernel(const unsigned short* __restrict__ qh,
                 const unsigned short* __restrict__ kh,
                 const unsigned short* __restrict__ vhT,
                 float* __restrict__ attn_out,
                 unsigned short* __restrict__ ho)               // [8][4096][128]
{
    __shared__ alignas(16) float          sP [16 * 1024];       // 64 KB scores
    __shared__ alignas(16) unsigned short sPb[16 * 1024];       // 32 KB probs bf16
    __shared__ float sred[256];
    __shared__ float srow[16];

    const int n = blockIdx.z, b = blockIdx.y, qt = blockIdx.x;
    const int wave = threadIdx.x >> 5, lane = threadIdx.x & 31;
    const int half = lane >> 4, nn = lane & 15;

    const unsigned short* Qh = qh + ((long)n * BQ + b * 1024 + qt * 16) * DK;
    const unsigned short* Kh = kh + ((long)n * BQ + b * 1024) * DK;   // [1024][128]

    // ---- phase 1: scores (scale pre-folded into qh) ----
    for (int i = 0; i < 8; ++i) {
        const int st = i * 8 + wave;                            // s-tile 0..63
        v8f acc[1][1]; acc[0][0] = vzero8();
        wmma_block_kloop<1, 1, true, true>(acc, Qh, DK,
                                           Kh + (long)st * 16 * DK, DK, DK);
#pragma unroll
        for (int r = 0; r < 8; ++r)
            sP[(r + 8 * half) * 1024 + st * 16 + nn] = acc[0][0][r];
    }
    __syncthreads();

    // ---- softmax over S=1024 per row (16 threads x 64 elems) ----
    const int row = threadIdx.x >> 4, chunk = threadIdx.x & 15;
    float* prow = sP + row * 1024 + chunk * 64;
    float mx = -1e30f;
    for (int j = 0; j < 64; ++j) mx = fmaxf(mx, prow[j]);
    sred[threadIdx.x] = mx;
    __syncthreads();
    if (chunk == 0) {
        float m2 = -1e30f;
        for (int j = 0; j < 16; ++j) m2 = fmaxf(m2, sred[row * 16 + j]);
        srow[row] = m2;
    }
    __syncthreads();
    mx = srow[row];
    float sum = 0.0f;
    for (int j = 0; j < 64; ++j) { float e = __expf(prow[j] - mx); prow[j] = e; sum += e; }
    __syncthreads();                       // everyone read srow before reuse
    sred[threadIdx.x] = sum;
    __syncthreads();
    if (chunk == 0) {
        float s2 = 0.0f;
        for (int j = 0; j < 16; ++j) s2 += sred[row * 16 + j];
        srow[row] = s2;
    }
    __syncthreads();
    const float inv = 1.0f / srow[row];

    // normalize: f32 -> global attn output, bf16 -> LDS for the P@V GEMM
    float* gattn = attn_out +
        (((long)n * 4 + b) * 1024 + qt * 16 + row) * 1024L + chunk * 64;
    unsigned short* pb16 = sPb + row * 1024 + chunk * 64;
    for (int j = 0; j < 64; j += 8) {
        float p[8];
        Frag t;
#pragma unroll
        for (int e = 0; e < 8; ++e) { p[e] = prow[j + e] * inv; t.u[e] = f2bf(p[e]); }
        float4 lo, hi;
        lo.x = p[0]; lo.y = p[1]; lo.z = p[2]; lo.w = p[3];
        hi.x = p[4]; hi.y = p[5]; hi.z = p[6]; hi.w = p[7];
        *(float4*)(gattn + j)     = lo;
        *(float4*)(gattn + j + 4) = hi;
        *(v4u*)(pb16 + j) = t.q[0];
    }
    __syncthreads();

    // ---- phase 2: out = P(16x1024) @ V(1024x128); wave owns one dk tile ----
    {
        const int d0 = wave * 16;
        const unsigned short* Vt = vhT + ((long)(n * 4 + b) * DK + d0) * 1024;
        v8f acc[1][1]; acc[0][0] = vzero8();
        wmma_block_kloop<1, 1, false, true>(acc, sPb, 1024, Vt, 1024, 1024);
        unsigned short* H = ho + ((long)n * BQ + b * 1024 + qt * 16) * DK;
#pragma unroll
        for (int r = 0; r < 8; ++r)
            H[(long)(r + 8 * half) * DK + d0 + nn] = f2bf(acc[0][0][r]);
    }
}

// ============ Kernel 3: w_o proj + k-gate + FC + SELU + a-gate head-sum
//              + residual + LayerNorm; 32-row tiles, 2x2 register blocking ===
__global__ __launch_bounds__(256)
void head_fc_kernel(const unsigned short* __restrict__ ho,     // [8][4096][128]
                    const unsigned short* __restrict__ woT,    // [8][2048][128]
                    const float* __restrict__ a_gate, const float* __restrict__ k_gate,
                    const unsigned short* __restrict__ fcT,    // [1024][2048]
                    const float* __restrict__ fc_b,
                    const float* __restrict__ residual,
                    const float* __restrict__ ln_g,  const float* __restrict__ ln_b,
                    float* __restrict__ yout)
{
    __shared__ alignas(16) unsigned short sT[32 * 2048]; // 128KB t(bf16) / f32 y
    __shared__ float sred[256], sred2[256];
    __shared__ float smu[32], svr[32];

    const int mt = blockIdx.x;                           // 32-row tile 0..127
    const int wave = threadIdx.x >> 5, lane = threadIdx.x & 31;
    const int half = lane >> 4, nn = lane & 15;
    const int m0 = mt * 32;

    // softmax of the 8-wide gates (cheap, per-thread redundant)
    float ag[NH], kg[NH];
    {
        float amx = -1e30f, kmx = -1e30f;
        for (int i = 0; i < NH; ++i) {
            ag[i] = a_gate[i]; kg[i] = k_gate[i];
            amx = fmaxf(amx, ag[i]); kmx = fmaxf(kmx, kg[i]);
        }
        float as = 0.0f, ks = 0.0f;
        for (int i = 0; i < NH; ++i) {
            ag[i] = __expf(ag[i] - amx); as += ag[i];
            kg[i] = __expf(kg[i] - kmx); ks += kg[i];
        }
        for (int i = 0; i < NH; ++i) { ag[i] /= as; kg[i] /= ks; }
    }

    v8f accy[4][2][2];                  // wave's 32x128 slice of the head-sum
#pragma unroll
    for (int t = 0; t < 4; ++t)
#pragma unroll
        for (int mi = 0; mi < 2; ++mi)
#pragma unroll
            for (int ni = 0; ni < 2; ++ni) accy[t][mi][ni] = vzero8();

    for (int n = 0; n < NH; ++n) {
        const unsigned short* Hn = ho + ((long)n * BQ + m0) * DK;
        __syncthreads();                // protect sT from previous head's readers

        // phase A: t(32x2048) = Hn(32x128) @ w_o[n](128x2048) * kg[n] -> bf16 LDS
        for (int i = 0; i < 8; ++i) {
            const int c0 = (i * 8 + wave) * 32;          // 64 col-groups of 32
            v8f acc[2][2];
#pragma unroll
            for (int mi = 0; mi < 2; ++mi)
#pragma unroll
                for (int ni = 0; ni < 2; ++ni) acc[mi][ni] = vzero8();
            wmma_block_kloop<2, 2, true, true>(acc, Hn, DK,
                woT + ((long)n * 2048 + c0) * DK, DK, DK);
#pragma unroll
            for (int mi = 0; mi < 2; ++mi)
#pragma unroll
                for (int ni = 0; ni < 2; ++ni)
#pragma unroll
                    for (int r = 0; r < 8; ++r)
                        sT[(mi * 16 + r + 8 * half) * 2048 + c0 + ni * 16 + nn]
                            = f2bf(acc[mi][ni][r] * kg[n]);
        }
        __syncthreads();

        // phase B: accy += selu(t @ fc_w + fc_b) * ag[n]
        for (int t = 0; t < 4; ++t) {
            const int c0 = (t * 8 + wave) * 32;          // 32 col-groups of 32
            v8f f[2][2];
#pragma unroll
            for (int mi = 0; mi < 2; ++mi)
#pragma unroll
                for (int ni = 0; ni < 2; ++ni) f[mi][ni] = vzero8();
            wmma_block_kloop<2, 2, false, true>(f, sT, 2048,
                fcT + (long)c0 * 2048, 2048, 2048);
#pragma unroll
            for (int mi = 0; mi < 2; ++mi)
#pragma unroll
                for (int ni = 0; ni < 2; ++ni) {
                    const float bcol = fc_b[c0 + ni * 16 + nn];
#pragma unroll
                    for (int r = 0; r < 8; ++r) {
                        const float x = f[mi][ni][r] + bcol;
                        const float s = (x > 0.0f) ? x
                                       : SELU_A * (__expf(x) - 1.0f);
                        accy[t][mi][ni][r] += SELU_S * s * ag[n];
                    }
                }
        }
    }

    // residual add into LDS (reuse sT as f32 32x1024)
    float* sY = (float*)sT;
    __syncthreads();
    const float* res = residual + (long)m0 * HSZ;
#pragma unroll
    for (int t = 0; t < 4; ++t)
#pragma unroll
        for (int mi = 0; mi < 2; ++mi)
#pragma unroll
            for (int ni = 0; ni < 2; ++ni)
#pragma unroll
                for (int r = 0; r < 8; ++r) {
                    const int rr = mi * 16 + r + 8 * half;
                    const int cc = (t * 8 + wave) * 32 + ni * 16 + nn;
                    sY[rr * HSZ + cc] = accy[t][mi][ni][r]
                                      + res[(long)rr * HSZ + cc];
                }
    __syncthreads();

    // LayerNorm over H=1024 per row (32 rows, 8 threads x 128 elems per row)
    const int row = threadIdx.x >> 3, chunk = threadIdx.x & 7;
    float* pr = sY + row * HSZ + chunk * 128;
    float s = 0.0f, s2 = 0.0f;
    for (int j = 0; j < 128; ++j) { const float x = pr[j]; s += x; s2 += x * x; }
    sred[threadIdx.x] = s; sred2[threadIdx.x] = s2;
    __syncthreads();
    if (chunk == 0) {
        float a = 0.0f, b2 = 0.0f;
        for (int j = 0; j < 8; ++j) { a += sred[row * 8 + j]; b2 += sred2[row * 8 + j]; }
        const float mu = a * (1.0f / HSZ);
        smu[row] = mu;
        svr[row] = b2 * (1.0f / HSZ) - mu * mu;
    }
    __syncthreads();
    const float mu   = smu[row];
    const float rstd = rsqrtf(svr[row] + 1e-6f);
    float* gy = yout + ((long)m0 + row) * HSZ + chunk * 128;
    const float* g  = ln_g + chunk * 128;
    const float* be = ln_b + chunk * 128;
    for (int j = 0; j < 128; ++j)
        gy[j] = (pr[j] - mu) * rstd * g[j] + be[j];
}

// ============================ launcher =====================================
extern "C" void kernel_launch(void* const* d_in, const int* in_sizes, int n_in,
                              void* d_out, int out_size, void* d_ws, size_t ws_size,
                              hipStream_t stream)
{
    const float* q      = (const float*)d_in[0];
    const float* k      = (const float*)d_in[1];
    const float* v      = (const float*)d_in[2];
    const float* w_q    = (const float*)d_in[3];
    const float* w_k    = (const float*)d_in[4];
    const float* w_v    = (const float*)d_in[5];
    const float* w_o    = (const float*)d_in[6];
    const float* a_gate = (const float*)d_in[7];
    const float* k_gate = (const float*)d_in[8];
    const float* fc_w   = (const float*)d_in[9];
    const float* fc_b   = (const float*)d_in[10];
    const float* ln_g   = (const float*)d_in[11];
    const float* ln_b   = (const float*)d_in[12];

    float* y_out    = (float*)d_out;                         // [4096,1024]
    float* attn_out = (float*)d_out + (size_t)BQ * HSZ;      // [32,1024,1024]

    // ---- workspace carve-up (all bf16 / unsigned short) ----
    unsigned short* xin = (unsigned short*)d_ws;             // [3][4096][1024]
    unsigned short* wt  = xin + (size_t)3 * BQ * HSZ;        // [3][8][128][1024]
    unsigned short* woT = wt  + (size_t)3 * NH * DK * HSZ;   // [8][2048][128]
    unsigned short* fcT = woT + (size_t)NH * 2048 * DK;      // [1024][2048]
    unsigned short* qh  = fcT + (size_t)HSZ * 2048;          // [8][4096][128]
    unsigned short* kh  = qh  + (size_t)NH * BQ * DK;
    unsigned short* vhT = kh  + (size_t)NH * BQ * DK;        // [8][4][128][1024]
    unsigned short* ho  = vhT + (size_t)NH * BQ * DK;        // [8][4096][128]

    // ---- prep: convert activations, transpose+convert weights ----
    {
        const long n8 = (long)BQ * HSZ / 8;                  // per activation array
        convert_bf16_kernel<<<(n8 + 255) / 256, 256, 0, stream>>>(
            (const float4*)q, (v4u*)xin, n8);
        convert_bf16_kernel<<<(n8 + 255) / 256, 256, 0, stream>>>(
            (const float4*)k, (v4u*)(xin + (size_t)BQ * HSZ), n8);
        convert_bf16_kernel<<<(n8 + 255) / 256, 256, 0, stream>>>(
            (const float4*)v, (v4u*)(xin + (size_t)2 * BQ * HSZ), n8);

        const long tw = (long)NH * HSZ * DK;                 // w_q/w_k/w_v each
        transpose_bf16_kernel<<<(tw / 2 + 255) / 256, 256, 0, stream>>>(
            w_q, wt,                             HSZ, DK, tw);
        transpose_bf16_kernel<<<(tw / 2 + 255) / 256, 256, 0, stream>>>(
            w_k, wt + (size_t)NH * DK * HSZ,     HSZ, DK, tw);
        transpose_bf16_kernel<<<(tw / 2 + 255) / 256, 256, 0, stream>>>(
            w_v, wt + (size_t)2 * NH * DK * HSZ, HSZ, DK, tw);

        const long to = (long)NH * DK * 2048;
        transpose_bf16_kernel<<<(to / 2 + 255) / 256, 256, 0, stream>>>(
            w_o, woT, DK, 2048, to);
        const long tf = (long)2048 * HSZ;
        transpose_bf16_kernel<<<(tf / 2 + 255) / 256, 256, 0, stream>>>(
            fc_w, fcT, 2048, HSZ, tf);
    }

    // ---- main pipeline ----
    qkv_proj_kernel<<<dim3(64, NH, 3), 256, 0, stream>>>(xin, wt, qh, kh, vhT);
    attn_kernel<<<dim3(64, 4, NH), 256, 0, stream>>>(qh, kh, vhT, attn_out, ho);
    head_fc_kernel<<<dim3(128), 256, 0, stream>>>(
        ho, woT, a_gate, k_gate, fcT, fc_b, q, ln_g, ln_b, y_out);
}
